// Model_46703474377349
// MI455X (gfx1250) — matrix operand order repaired and checked by reading, compiled, not run on hardware
//
#include <hip/hip_runtime.h>

typedef __bf16 bf16;
typedef __attribute__((ext_vector_type(16))) __bf16 v16bf;
typedef __attribute__((ext_vector_type(8)))  __bf16 v8bf;
typedef __attribute__((ext_vector_type(8)))  float  v8f;

#define HGN  152   // hidden size
#define NPAD 160   // padded output cols (10 x 16)
#define NT   10    // N tiles
#define FE   76    // edge/input feature size

// ---------------- helper kernels ----------------

__global__ void k_cvt_bf16(const float* __restrict__ in, bf16* __restrict__ out, int n) {
  int i = blockIdx.x * 256 + threadIdx.x;
  if (i < n) out[i] = (bf16)in[i];
}

__global__ void k_zero(float* __restrict__ p, int n) {
  int i = blockIdx.x * 256 + threadIdx.x;
  if (i < n) p[i] = 0.0f;
}

// Swizzle fp32 weight [K,152] into per-lane WMMA-B fragment order (bf16):
//   flat[(((nt*nkc)+kc)*32 + lane)*16 + j]
//   k = kc*32 + (lane>>4)*16 + j , n = nt*16 + (lane&15)
// so each lane's 16-element fragment is one contiguous 32-byte load.
__global__ void k_padw_swz(const float* __restrict__ w, int K,
                           bf16* __restrict__ out, int nkc) {
  int i = blockIdx.x * 256 + threadIdx.x;
  if (i >= NT * nkc * 32 * 16) return;
  int j    = i & 15;
  int lane = (i >> 4) & 31;
  int rest = i >> 9;
  int kc   = rest % nkc;
  int nt   = rest / nkc;
  int k = kc * 32 + (lane >> 4) * 16 + j;
  int n = nt * 16 + (lane & 15);
  out[i] = (k < K && n < HGN) ? (bf16)w[k * HGN + n] : (bf16)0.0f;
}

__global__ void k_deg(const int* __restrict__ dst, float* __restrict__ deg, int n_edges) {
  int e = blockIdx.x * 256 + threadIdx.x;
  if (e < n_edges) unsafeAtomicAdd(&deg[dst[e]], 1.0f);
}

__global__ void k_mean(float* __restrict__ agg, const float* __restrict__ deg, int total) {
  int i = blockIdx.x * 256 + threadIdx.x;
  if (i < total) agg[i] = agg[i] / fmaxf(deg[i / HGN], 1.0f);
}

// ---------------- shared device helpers ----------------

// stage one A row (KPAD cols) into LDS with 16B vector stores; 2 threads per row.
// first `ln` cols from bf16 row, next `ln2` cols from fp32 row, rest zero.
template<int KPAD>
__device__ __forceinline__ void stage_a_row(bf16* __restrict__ arow,
                                            const bf16* __restrict__ nrow, int ln,
                                            const float* __restrict__ frow, int ln2,
                                            int tid) {
  constexpr int CPR = KPAD >> 3;  // 16B chunks per row
  for (int cc = (tid & 1); cc < CPR; cc += 2) {
    int c0 = cc * 8;
    v8bf buf;
    #pragma unroll
    for (int j = 0; j < 8; ++j) {
      int c = c0 + j;
      bf16 v;
      if (c < ln)            v = nrow[c];
      else if (c < ln + ln2) v = (bf16)frow[c - ln];
      else                   v = (bf16)0.0f;
      buf[j] = v;
    }
    *(v8bf*)(arow + c0) = buf;
  }
}

template<int KPAD>
__device__ __forceinline__ void stage_a_zero(bf16* __restrict__ arow, int tid) {
  constexpr int CPR = KPAD >> 3;
  v8bf z;
  #pragma unroll
  for (int j = 0; j < 8; ++j) z[j] = (bf16)0.0f;
  for (int cc = (tid & 1); cc < CPR; cc += 2) *(v8bf*)(arow + cc * 8) = z;
}

// compute one wave's 16xNPAD tile: A from LDS (2 x ds_load_b128 per K-chunk),
// B from global at compile-time immediate offsets, WMMA accumulate.
template<int KPAD>
__device__ __forceinline__ void wmma_tile(const bf16* __restrict__ A_lds,
                                          const bf16* __restrict__ Wp,
                                          int wave, int lane, v8f acc[NT]) {
  constexpr int NKC = KPAD >> 5;
  const int lhi = lane >> 4, lm = lane & 15;
  #pragma unroll
  for (int i = 0; i < NT; ++i)
    #pragma unroll
    for (int j = 0; j < 8; ++j) acc[i][j] = 0.0f;

  // per-lane fragment base: element offset lane*16; kc advances by 512 elems
  const v16bf* __restrict__ Wf = (const v16bf*)Wp + lane;
  for (int kc = 0; kc < NKC; ++kc) {
    const v8bf* ap = (const v8bf*)(A_lds + (wave * 16 + lm) * KPAD + kc * 32 + lhi * 16);
    v8bf alo = ap[0];
    v8bf ahi = ap[1];
    v16bf a = __builtin_shufflevector(alo, ahi,
                                      0, 1, 2, 3, 4, 5, 6, 7,
                                      8, 9, 10, 11, 12, 13, 14, 15);
    const v16bf* bkc = Wf + kc * 32;      // one address register per kc
    #pragma unroll
    for (int nt = 0; nt < NT; ++nt) {
      v16bf b = bkc[nt * NKC * 32];       // compile-time immediate offset
      acc[nt] = __builtin_amdgcn_wmma_f32_16x16x32_bf16(
          false, a, false, b, (short)0, acc[nt], false, false);
    }
  }
}

// ---------------- message GEMM: msg = [h[src] | efeat] @ W + b, scatter-add to agg[dst] ----------------
// block = 128 threads (4 waves), 64 edges/block, each wave owns a 16xNPAD tile.
template<int KPAD>
__global__ void k_msg_wmma(const bf16* __restrict__ nodef, int ln,
                           const float* __restrict__ efeats,
                           const int* __restrict__ src, const int* __restrict__ dst,
                           const bf16* __restrict__ Wp,
                           const float* __restrict__ bias,
                           float* __restrict__ agg, int n_edges)
{
  extern __shared__ char smem[];
  bf16* A_lds = (bf16*)smem;  // [64][KPAD]
  const int tid = threadIdx.x, wave = tid >> 5, lane = tid & 31;
  const int lhi = lane >> 4, lm = lane & 15;
  const int eBase = blockIdx.x * 64;

  { // stage A: 2 threads per row
    int r = tid >> 1;
    int e = eBase + r;
    bf16* arow = A_lds + r * KPAD;
    if (e < n_edges)
      stage_a_row<KPAD>(arow, nodef + (size_t)src[e] * ln, ln,
                        efeats + (size_t)e * FE, FE, tid);
    else
      stage_a_zero<KPAD>(arow, tid);
  }
  __syncthreads();

  v8f acc[NT];
  wmma_tile<KPAD>(A_lds, Wp, wave, lane, acc);

  // scatter-add into agg[dst] (C/D layout: lane = N, vgpr v -> M = lhi*8+v)
  int drow[8];
  #pragma unroll
  for (int v = 0; v < 8; ++v) {
    int e = eBase + wave * 16 + lhi * 8 + v;
    drow[v] = (e < n_edges) ? dst[e] : -1;
  }
  #pragma unroll
  for (int nt = 0; nt < NT; ++nt) {
    int n0 = nt * 16 + lm;
    if (n0 < HGN) {
      float bv = bias[n0];
      #pragma unroll
      for (int v = 0; v < 8; ++v)
        if (drow[v] >= 0)
          unsafeAtomicAdd(agg + (size_t)drow[v] * HGN + n0, acc[nt][v] + bv);
    }
  }
}

// ---------------- apply GEMM: h = relu([nf | h_neigh] @ W + b) -> bf16 ----------------
template<int KPAD>
__global__ void k_apply_wmma(const bf16* __restrict__ nf, int ln,
                             const float* __restrict__ hneigh,
                             const bf16* __restrict__ Wp,
                             const float* __restrict__ bias,
                             bf16* __restrict__ hout, int n_nodes)
{
  extern __shared__ char smem[];
  bf16* A_lds = (bf16*)smem;
  const int tid = threadIdx.x, wave = tid >> 5, lane = tid & 31;
  const int lhi = lane >> 4, lm = lane & 15;
  const int nBase = blockIdx.x * 64;

  {
    int r = tid >> 1;
    int node = nBase + r;
    bf16* arow = A_lds + r * KPAD;
    if (node < n_nodes)
      stage_a_row<KPAD>(arow, nf + (size_t)node * ln, ln,
                        hneigh + (size_t)node * HGN, HGN, tid);
    else
      stage_a_zero<KPAD>(arow, tid);
  }
  __syncthreads();

  v8f acc[NT];
  wmma_tile<KPAD>(A_lds, Wp, wave, lane, acc);

  #pragma unroll
  for (int nt = 0; nt < NT; ++nt) {
    int n0 = nt * 16 + lm;
    if (n0 < HGN) {
      float bv = bias[n0];
      #pragma unroll
      for (int v = 0; v < 8; ++v) {
        int node = nBase + wave * 16 + lhi * 8 + v;
        if (node < n_nodes)
          hout[(size_t)node * HGN + n0] = (bf16)fmaxf(acc[nt][v] + bv, 0.0f);
      }
    }
  }
}

// ---------------- predictor: score = [h[src] | h[dst]] @ w_pred + b_pred ----------------
__global__ void k_pred(const bf16* __restrict__ h, const int* __restrict__ src,
                       const int* __restrict__ dst, const float* __restrict__ w,
                       const float* __restrict__ b, float* __restrict__ out, int n_edges)
{
  __shared__ float wl[2 * HGN * 2];
  for (int i = threadIdx.x; i < 2 * HGN * 2; i += 256) wl[i] = w[i];
  __syncthreads();
  int e = blockIdx.x * 256 + threadIdx.x;
  if (e >= n_edges) return;
  const bf16* hs = h + (size_t)src[e] * HGN;
  const bf16* hd = h + (size_t)dst[e] * HGN;
  float a0 = b[0], a1 = b[1];
  for (int j = 0; j < HGN; ++j) {
    float v = (float)hs[j];
    a0 += v * wl[j * 2 + 0];
    a1 += v * wl[j * 2 + 1];
  }
  for (int j = 0; j < HGN; ++j) {
    float v = (float)hd[j];
    a0 += v * wl[(HGN + j) * 2 + 0];
    a1 += v * wl[(HGN + j) * 2 + 1];
  }
  out[(size_t)e * 2 + 0] = a0;
  out[(size_t)e * 2 + 1] = a1;
}

// ---------------- host orchestration ----------------

extern "C" void kernel_launch(void* const* d_in, const int* in_sizes, int n_in,
                              void* d_out, int out_size, void* d_ws, size_t ws_size,
                              hipStream_t stream) {
  const float* nfeats  = (const float*)d_in[0];
  const float* efeats  = (const float*)d_in[1];
  const float* w_msg1  = (const float*)d_in[2];
  const float* b_msg1  = (const float*)d_in[3];
  const float* w_app1  = (const float*)d_in[4];
  const float* b_app1  = (const float*)d_in[5];
  const float* w_msg2  = (const float*)d_in[6];
  const float* b_msg2  = (const float*)d_in[7];
  const float* w_app2  = (const float*)d_in[8];
  const float* b_app2  = (const float*)d_in[9];
  const float* w_msg3  = (const float*)d_in[10];
  const float* b_msg3  = (const float*)d_in[11];
  const float* w_app3  = (const float*)d_in[12];
  const float* b_app3  = (const float*)d_in[13];
  const float* w_pred  = (const float*)d_in[14];
  const float* b_pred  = (const float*)d_in[15];
  const int*   src     = (const int*)d_in[16];
  const int*   dst     = (const int*)d_in[17];

  const int E = in_sizes[16];
  const int N = in_sizes[0] / FE;

  // workspace carve-up (256B aligned)
  char* ws = (char*)d_ws;
  size_t off = 0;
  auto take = [&](size_t bytes) -> char* {
    char* p = ws + off;
    off += (bytes + 255) & ~(size_t)255;
    return p;
  };
  bf16*  nf_bf = (bf16*)take((size_t)N * FE * sizeof(bf16));
  bf16*  hA    = (bf16*)take((size_t)N * HGN * sizeof(bf16));
  bf16*  hB    = (bf16*)take((size_t)N * HGN * sizeof(bf16));
  float* agg   = (float*)take((size_t)N * HGN * sizeof(float));
  float* deg   = (float*)take((size_t)N * sizeof(float));
  bf16*  wm1   = (bf16*)take((size_t)160 * NPAD * sizeof(bf16));
  bf16*  wm2   = (bf16*)take((size_t)256 * NPAD * sizeof(bf16));
  bf16*  wm3   = (bf16*)take((size_t)256 * NPAD * sizeof(bf16));
  bf16*  wa1   = (bf16*)take((size_t)256 * NPAD * sizeof(bf16));
  bf16*  wa2   = (bf16*)take((size_t)320 * NPAD * sizeof(bf16));
  bf16*  wa3   = (bf16*)take((size_t)320 * NPAD * sizeof(bf16));

  auto cdiv = [](int a, int b) { return (a + b - 1) / b; };

  // preconvert features; swizzle+pad weights to fragment-order bf16
  k_cvt_bf16<<<cdiv(N * FE, 256), 256, 0, stream>>>(nfeats, nf_bf, N * FE);
  k_padw_swz<<<cdiv(160 * NPAD, 256), 256, 0, stream>>>(w_msg1, FE + FE,   wm1, 160 / 32);
  k_padw_swz<<<cdiv(256 * NPAD, 256), 256, 0, stream>>>(w_msg2, HGN + FE,  wm2, 256 / 32);
  k_padw_swz<<<cdiv(256 * NPAD, 256), 256, 0, stream>>>(w_msg3, HGN + FE,  wm3, 256 / 32);
  k_padw_swz<<<cdiv(256 * NPAD, 256), 256, 0, stream>>>(w_app1, FE + HGN,  wa1, 256 / 32);
  k_padw_swz<<<cdiv(320 * NPAD, 256), 256, 0, stream>>>(w_app2, HGN + HGN, wa2, 320 / 32);
  k_padw_swz<<<cdiv(320 * NPAD, 256), 256, 0, stream>>>(w_app3, HGN + HGN, wa3, 320 / 32);

  // in-degrees (same for every layer)
  k_zero<<<cdiv(N, 256), 256, 0, stream>>>(deg, N);
  k_deg<<<cdiv(E, 256), 256, 0, stream>>>(dst, deg, E);

  // ---- layer 1 ----
  k_zero<<<cdiv(N * HGN, 256), 256, 0, stream>>>(agg, N * HGN);
  k_msg_wmma<160><<<cdiv(E, 64), 128, (size_t)64 * 160 * 2, stream>>>(
      nf_bf, FE, efeats, src, dst, wm1, b_msg1, agg, E);
  k_mean<<<cdiv(N * HGN, 256), 256, 0, stream>>>(agg, deg, N * HGN);
  k_apply_wmma<256><<<cdiv(N, 64), 128, (size_t)64 * 256 * 2, stream>>>(
      nf_bf, FE, agg, wa1, b_app1, hA, N);

  // ---- layer 2 ----
  k_zero<<<cdiv(N * HGN, 256), 256, 0, stream>>>(agg, N * HGN);
  k_msg_wmma<256><<<cdiv(E, 64), 128, (size_t)64 * 256 * 2, stream>>>(
      hA, HGN, efeats, src, dst, wm2, b_msg2, agg, E);
  k_mean<<<cdiv(N * HGN, 256), 256, 0, stream>>>(agg, deg, N * HGN);
  k_apply_wmma<320><<<cdiv(N, 64), 128, (size_t)64 * 320 * 2, stream>>>(
      hA, HGN, agg, wa2, b_app2, hB, N);

  // ---- layer 3 ----
  k_zero<<<cdiv(N * HGN, 256), 256, 0, stream>>>(agg, N * HGN);
  k_msg_wmma<256><<<cdiv(E, 64), 128, (size_t)64 * 256 * 2, stream>>>(
      hB, HGN, efeats, src, dst, wm3, b_msg3, agg, E);
  k_mean<<<cdiv(N * HGN, 256), 256, 0, stream>>>(agg, deg, N * HGN);
  k_apply_wmma<320><<<cdiv(N, 64), 128, (size_t)64 * 320 * 2, stream>>>(
      hB, HGN, agg, wa3, b_app3, hA, N);

  // ---- edge predictor ----
  k_pred<<<cdiv(E, 256), 256, 0, stream>>>(hA, src, dst, w_pred, b_pred,
                                           (float*)d_out, E);
}
